// iTCL_12601434047136
// MI455X (gfx1250) — matrix operand order repaired and checked by reading, not compiled, not run on hardware
//
#include <hip/hip_runtime.h>

typedef __attribute__((ext_vector_type(16))) _Float16 v16h;
typedef __attribute__((ext_vector_type(8)))  _Float16 v8h;
typedef __attribute__((ext_vector_type(8)))  float    v8f;
typedef __attribute__((ext_vector_type(4)))  int      v4i;

#define AS1 __attribute__((address_space(1)))
#define AS3 __attribute__((address_space(3)))

#if defined(__AMDGCN__) && \
    __has_builtin(__builtin_amdgcn_global_load_async_to_lds_b128) && \
    __has_builtin(__builtin_amdgcn_s_wait_asynccnt)
#define USE_ASYNC 1
#else
#define USE_ASYNC 0
#endif

namespace {
constexpr int kN    = 1024;
constexpr int kD    = 128;
constexpr int W1STR = 136;  // padded row stride (halfs), 16B-aligned rows
constexpr int W2STR = 128;
constexpr int HSTR  = 136;
constexpr int H3STR = 68;

union Frag { v16h v; v8h h[2]; };

// leaky_relu(x) == max(x, 0.01*x): lowers to dual-issued mul+max,
// no v_cmp/v_cndmask SGPR serialization.
__device__ inline float leaky(float x) { return __builtin_fmaxf(x, 0.01f * x); }
}

__global__ __launch_bounds__(128)
void cumsum_mlp_wmma(const float* __restrict__ hid,
                     const float* __restrict__ W1, const float* __restrict__ b1,
                     const float* __restrict__ W2, const float* __restrict__ b2,
                     const float* __restrict__ W3, const float* __restrict__ b3,
                     float* __restrict__ out)
{
    __shared__ _Float16 lw1[128 * W1STR];   // W1 row-major f16
    __shared__ _Float16 lw2[64  * W2STR];   // W2 row-major f16
    __shared__ _Float16 lhf[16 * HSTR];     // cumsum tile (A of layer 1)
    __shared__ _Float16 lh2[16 * HSTR];     // layer-1 activations (A of layer 2)
    __shared__ float    lh3[16 * H3STR];    // layer-2 activations (f32)
    __shared__ float    lb1[128];
    __shared__ float    lb2[64];
    __shared__ float    lw3[64];
#if USE_ASYNC
    __shared__ float    lraw[2][16 * kD];   // ping-pong raw f32 input tiles
#endif

    const int tid  = threadIdx.x;       // 0..127
    const int lane = tid & 31;
    const int wave = tid >> 5;          // 0..3
    const int b    = blockIdx.x;        // batch element

    // ---- stage weights/biases to LDS (f16 for WMMA B operands) ----
    #pragma unroll 4
    for (int r = 0; r < 128; ++r) lw1[r * W1STR + tid] = (_Float16)W1[r * kD + tid];
    #pragma unroll 4
    for (int r = 0; r < 64; ++r)  lw2[r * W2STR + tid] = (_Float16)W2[r * kD + tid];
    lb1[tid] = b1[tid];
    if (tid < 64) { lb2[tid] = b2[tid]; lw3[tid] = W3[tid]; }
    const float b3v = b3[0];
    __syncthreads();

    const int m16 = lane & 15;
    const int kh  = lane >> 4;          // 0 or 1 (half-wave)
    const int mb  = kh * 8;             // C/D row base for this half-wave

    // ---- hoist loop-invariant LDS values into registers ----
    const float bias1a = lb1[wave * 32 + m16];
    const float bias1b = lb1[wave * 32 + 16 + m16];
    const float bias2  = lb2[wave * 16 + m16];
    float w3r[8];
    #pragma unroll
    for (int j = 0; j < 8; ++j) w3r[j] = lw3[(tid & 7) * 8 + j];

    const float* hbase = hid + (size_t)b * kN * kD;
    float run = 0.f;                    // f32 running cumsum, thread t owns dim t

#if USE_ASYNC
    // ---- prologue: async-load tile 0 into lraw[0]; wave w owns rows 4w..4w+3 ----
    #pragma unroll
    for (int i = 0; i < 4; ++i) {
        const int row = wave * 4 + i;
        const float* s = hbase + (size_t)row * kD + lane * 4;   // 16B per lane
        float* d = &lraw[0][row * kD + lane * 4];
        __builtin_amdgcn_global_load_async_to_lds_b128((AS1 v4i*)s, (AS3 v4i*)d, 0, 0);
    }
    int cur = 0;
#else
    float v[16];
    #pragma unroll
    for (int r = 0; r < 16; ++r) v[r] = hbase[(size_t)r * kD + tid];
#endif

    for (int t = 0; t < kN / 16; ++t) {
        const int n0 = t * 16;

#if USE_ASYNC
        __builtin_amdgcn_s_wait_asynccnt(0);   // my rows of tile t landed
        __syncthreads();                       // all waves' rows visible
        #pragma unroll
        for (int r = 0; r < 16; ++r) {
            run += lraw[cur][r * kD + tid];
            lhf[r * HSTR + tid] = (_Float16)run;
        }
        if (t + 1 < kN / 16) {                 // kick off tile t+1 during GEMMs
            #pragma unroll
            for (int i = 0; i < 4; ++i) {
                const int row = wave * 4 + i;
                const float* s = hbase + (size_t)(n0 + 16 + row) * kD + lane * 4;
                float* d = &lraw[cur ^ 1][row * kD + lane * 4];
                __builtin_amdgcn_global_load_async_to_lds_b128((AS1 v4i*)s, (AS3 v4i*)d, 0, 0);
            }
        }
        cur ^= 1;
#else
        #pragma unroll
        for (int r = 0; r < 16; ++r) { run += v[r]; lhf[r * HSTR + tid] = (_Float16)run; }
        if (t + 1 < kN / 16) {                 // register prefetch of tile t+1
            #pragma unroll
            for (int r = 0; r < 16; ++r) v[r] = hbase[(size_t)(n0 + 16 + r) * kD + tid];
        }
#endif
        __syncthreads();

        // ---- layer 1: [16x128] @ W1^T -> [16x128], 2 col-tiles per wave ----
        #pragma unroll
        for (int cbi = 0; cbi < 2; ++cbi) {
            const int n0c = (wave * 2 + cbi) * 16;
            const float bias = cbi ? bias1b : bias1a;
            v8f acc;
            #pragma unroll
            for (int r = 0; r < 8; ++r) acc[r] = bias;   // bias preloaded as C
            #pragma unroll
            for (int ks = 0; ks < 4; ++ks) {
                const int k0 = ks * 32;
                Frag a, w;
                // A frag: lanes 0-15 hold K {0..7,16..23}, lanes 16-31 {8..15,24..31}
                const _Float16* ap = &lhf[m16 * HSTR + k0 + kh * 8];
                a.h[0] = *(const v8h*)(ap);
                a.h[1] = *(const v8h*)(ap + 16);
                // B frag: column n = n0c+lane%16 == W1 row; same K split as A
                const _Float16* bp = &lw1[(n0c + m16) * W1STR + k0 + kh * 8];
                w.h[0] = *(const v8h*)(bp);
                w.h[1] = *(const v8h*)(bp + 16);
                acc = __builtin_amdgcn_wmma_f32_16x16x32_f16(false, a.v, false, w.v,
                                                             (short)0, acc, false, false);
            }
            #pragma unroll
            for (int r = 0; r < 8; ++r)
                lh2[(mb + r) * HSTR + n0c + m16] = (_Float16)leaky(acc[r]);
        }
        __syncthreads();

        // ---- layer 2: [16x128] @ W2^T -> [16x64], 1 col-tile per wave ----
        {
            const int n0c = wave * 16;
            v8f acc;
            #pragma unroll
            for (int r = 0; r < 8; ++r) acc[r] = bias2;
            #pragma unroll
            for (int ks = 0; ks < 4; ++ks) {
                const int k0 = ks * 32;
                Frag a, w;
                const _Float16* ap = &lh2[m16 * HSTR + k0 + kh * 8];
                a.h[0] = *(const v8h*)(ap);
                a.h[1] = *(const v8h*)(ap + 16);
                const _Float16* bp = &lw2[(n0c + m16) * W2STR + k0 + kh * 8];
                w.h[0] = *(const v8h*)(bp);
                w.h[1] = *(const v8h*)(bp + 16);
                acc = __builtin_amdgcn_wmma_f32_16x16x32_f16(false, a.v, false, w.v,
                                                             (short)0, acc, false, false);
            }
            #pragma unroll
            for (int r = 0; r < 8; ++r)
                lh3[(mb + r) * H3STR + n0c + m16] = leaky(acc[r]);  // f32 for final dot
        }
        __syncthreads();

        // ---- layer 3: [16x64] . w3 -> 16 scalars (VALU + shuffle reduce) ----
        {
            const int m = tid >> 3;            // output row 0..15
            const int s = (tid & 7) * 8;       // 8-element chunk
            float p = 0.f;
            #pragma unroll
            for (int j = 0; j < 8; ++j) p += lh3[m * H3STR + s + j] * w3r[j];
            p += __shfl_xor(p, 1, 8);
            p += __shfl_xor(p, 2, 8);
            p += __shfl_xor(p, 4, 8);
            const int n = n0 + m;
            if ((tid & 7) == 0 && n > 0)       // cumsum[:,1:,:] -> drop n==0
                __builtin_nontemporal_store(p + b3v, &out[(size_t)b * (kN - 1) + (n - 1)]);
        }
    }
}

extern "C" void kernel_launch(void* const* d_in, const int* in_sizes, int n_in,
                              void* d_out, int out_size, void* d_ws, size_t ws_size,
                              hipStream_t stream) {
    (void)in_sizes; (void)n_in; (void)d_ws; (void)ws_size; (void)out_size;
    const float* hid = (const float*)d_in[0];
    const float* W1  = (const float*)d_in[1];
    const float* b1v = (const float*)d_in[2];
    const float* W2  = (const float*)d_in[3];
    const float* b2v = (const float*)d_in[4];
    const float* W3  = (const float*)d_in[5];
    const float* b3v = (const float*)d_in[6];
    float* out = (float*)d_out;
    hipLaunchKernelGGL(cumsum_mlp_wmma, dim3(512), dim3(128), 0, stream,
                       hid, W1, b1v, W2, b2v, W3, b3v, out);
}
